// Encoder_10316511445501
// MI455X (gfx1250) — matrix operand order repaired and checked by reading, not compiled, not run on hardware
//
#include <hip/hip_runtime.h>
#include <hip/hip_bf16.h>
#include <math.h>

typedef unsigned short u16;
typedef __attribute__((ext_vector_type(16))) __bf16 v16bf;
typedef __attribute__((ext_vector_type(8)))  float  v8f;
typedef __attribute__((ext_vector_type(4)))  int    i32x4;

#define B_SAMP 4096
#define XDIM   1024
#define HDIM   1024
#define MU_N   512      // N_CLASSES*N_ENVS*D2 = 4*128
#define D2     128
#define NT     8256     // 128*129/2
#define COV_N  33024    // 4*NT
#define LDP    40       // padded LDS row stride (bf16), 80B = 16B aligned, conflict-free
#define BUFSZ  (128 * LDP)

#ifndef __has_builtin
#define __has_builtin(x) 0
#endif

#if __has_builtin(__builtin_amdgcn_global_load_async_to_lds_b128)
#define HAVE_ASYNC 1
#else
#define HAVE_ASYNC 0
#endif

union FragU { uint4 q[2]; v16bf v; };

__device__ __forceinline__ u16 f2bf(float f) {
    unsigned u = __float_as_uint(f);
    unsigned r = (u + 0x7FFFu + ((u >> 16) & 1u)) >> 16;
    return (u16)r;
}

#if HAVE_ASYNC
__device__ __forceinline__ void async_cp16(const u16* g, u16* l) {
    // builtin signature (from hipcc diagnostic): (AS1 i32x4*, AS3 i32x4*, imm offset, imm cpol)
    i32x4* gm = (i32x4*)const_cast<u16*>(g);
    i32x4* lm = (i32x4*)l;
    __builtin_amdgcn_global_load_async_to_lds_b128(
        (__attribute__((address_space(1))) i32x4*)gm,
        (__attribute__((address_space(3))) i32x4*)lm, 0, 0);
}
__device__ __forceinline__ void wait_async0() {
#if __has_builtin(__builtin_amdgcn_s_wait_asynccnt)
    __builtin_amdgcn_s_wait_asynccnt(0);
#else
    asm volatile("s_wait_asynccnt 0x0" ::: "memory");
#endif
}
__device__ __forceinline__ void wait_async4() {
#if __has_builtin(__builtin_amdgcn_s_wait_asynccnt)
    __builtin_amdgcn_s_wait_asynccnt(4);
#else
    asm volatile("s_wait_asynccnt 0x4" ::: "memory");
#endif
}
#endif

// ---------------- utility kernels ----------------

__global__ void cvt_f32_bf16(const float* __restrict__ in, u16* __restrict__ out, long n) {
    long i = (long)blockIdx.x * blockDim.x + threadIdx.x;
    if (i < n) out[i] = f2bf(in[i]);
}

// in: [rows, cols] f32 -> out: [cols, rows] bf16 (tiled 32x32 transpose; dims % 32 == 0)
__global__ __launch_bounds__(256)
void cvt_transpose_bf16(const float* __restrict__ in, u16* __restrict__ out,
                        int rows, int cols) {
    __shared__ float tile[32][33];
    const int bc = blockIdx.x * 32;          // col base
    const int br = blockIdx.y * 32;          // row base
    const int tx = threadIdx.x & 31;
    const int ty = threadIdx.x >> 5;         // 0..7
    for (int i = 0; i < 32; i += 8)
        tile[ty + i][tx] = in[(size_t)(br + ty + i) * cols + (bc + tx)];
    __syncthreads();
    for (int i = 0; i < 32; i += 8)
        out[(size_t)(bc + ty + i) * rows + (br + tx)] = f2bf(tile[tx][ty + i]);
}

__global__ void zero_f32v4(float* __restrict__ p, long n4) {
    long i = (long)blockIdx.x * blockDim.x + threadIdx.x;
    if (i < n4) ((float4*)p)[i] = make_float4(0.f, 0.f, 0.f, 0.f);
}

__global__ void zero_cnt(int* cnt) {
    if (threadIdx.x < 4) cnt[threadIdx.x] = 0;
}

__global__ void build_groups(const int* __restrict__ y, const int* __restrict__ e,
                             int* __restrict__ cnt, int* __restrict__ rows) {
    int s = blockIdx.x * blockDim.x + threadIdx.x;
    if (s < B_SAMP) {
        int g = y[s] * 2 + e[s];
        int pos = atomicAdd(&cnt[g], 1);
        rows[g * B_SAMP + pos] = s;
    }
}

// ---------------- WMMA compute step ----------------
// sAbuf: activation tile [128 rows][32 k]   (row stride LDP)
// sBbuf: weight-T tile   [128 cols][32 k]   (row stride LDP)
__device__ __forceinline__ void compute_tile(const u16* __restrict__ sAbuf,
                                             const u16* __restrict__ sBbuf,
                                             int waveM, int waveN, int h, int ln,
                                             v8f acc[2][4]) {
    FragU a0, a1;
    const int m0 = waveM * 32 + ln;
    const int m1 = m0 + 16;
    a0.q[0] = *(const uint4*)(&sAbuf[m0 * LDP + 8 * h]);        // K = 8h..8h+7
    a0.q[1] = *(const uint4*)(&sAbuf[m0 * LDP + 16 + 8 * h]);   // K = 16+8h..23+8h
    a1.q[0] = *(const uint4*)(&sAbuf[m1 * LDP + 8 * h]);
    a1.q[1] = *(const uint4*)(&sAbuf[m1 * LDP + 16 + 8 * h]);
    for (int j = 0; j < 4; ++j) {
        const int n = waveN * 64 + j * 16 + ln;
        FragU b;
        b.q[0] = *(const uint4*)(&sBbuf[n * LDP + 8 * h]);      // K = 8h..8h+7
        b.q[1] = *(const uint4*)(&sBbuf[n * LDP + 16 + 8 * h]); // K = 16+8h..23+8h
        acc[0][j] = __builtin_amdgcn_wmma_f32_16x16x32_bf16(
            false, a0.v, false, b.v, (short)0, acc[0][j], false, false);
        acc[1][j] = __builtin_amdgcn_wmma_f32_16x16x32_bf16(
            false, a1.v, false, b.v, (short)0, acc[1][j], false, false);
    }
}

// ---------------- WMMA GEMM ----------------
// A  : activations [M, K] bf16 row-major
// Bt : weights transposed [N, K] bf16 row-major
// EPI 0: relu(A@B + bias) -> bf16 ; EPI 1: mu head masked f32 store ;
// EPI 2: cov head grouped rows, scatter into L with softplus diagonal
template <int EPI>
__global__ __launch_bounds__(256)
void gemm_wmma(const u16* __restrict__ A, const u16* __restrict__ Bt,
               const float* __restrict__ bias,
               u16* __restrict__ outB, float* __restrict__ outF,
               int M, int N, int K,
               const int* __restrict__ yv, const int* __restrict__ ev,
               const int* __restrict__ grpCnt, const int* __restrict__ grpRows)
{
    const int tid   = threadIdx.x;
    const int lane  = tid & 31;
    const int wave  = tid >> 5;
    const int waveM = wave & 3;
    const int waveN = wave >> 2;
    const int h     = lane >> 4;
    const int ln    = lane & 15;

    const int tileN = blockIdx.x;
    int g = 0, rowTile = blockIdx.y, cnt = M;
    if (EPI == 2) {
        g       = blockIdx.y >> 5;
        rowTile = blockIdx.y & 31;
        cnt     = grpCnt[g];
        if (rowTile * 128 >= cnt) return;
    }

    __shared__ u16 sA[2][BUFSZ];
    __shared__ u16 sB[2][BUFSZ];
    __shared__ int sRows[128];

    for (int i = tid; i < 128; i += 256) {
        int r = rowTile * 128 + i;
        if (EPI == 2) {
            int rc = (r < cnt) ? r : (cnt - 1);
            sRows[i] = grpRows[g * B_SAMP + rc];
        } else {
            sRows[i] = r;
        }
    }
    __syncthreads();

    // per-thread staging slot: 2 threads per tile-row, 32 bytes each
    const int row  = tid >> 1;
    const int half = tid & 1;
    const u16* gA = A + (size_t)sRows[row] * (size_t)K + (size_t)(half * 16);
    size_t bcol;
    if (EPI == 2) {
        size_t c = (size_t)tileN * 128 + row;
        if (c > (size_t)(NT - 1)) c = NT - 1;
        bcol = (size_t)g * NT + c;
    } else {
        bcol = (size_t)tileN * 128 + row;
    }
    const u16* gB = Bt + bcol * (size_t)K + (size_t)(half * 16);
    u16* lA = &sA[0][row * LDP + half * 16];
    u16* lB = &sB[0][row * LDP + half * 16];

    v8f acc[2][4];
    for (int i = 0; i < 2; ++i)
        for (int j = 0; j < 4; ++j)
            acc[i][j] = (v8f){0.f,0.f,0.f,0.f,0.f,0.f,0.f,0.f};

#if HAVE_ASYNC
    auto issue = [&](int kt, int buf) {
        async_cp16(gA + kt,     lA + buf * BUFSZ);
        async_cp16(gA + kt + 8, lA + buf * BUFSZ + 8);
        async_cp16(gB + kt,     lB + buf * BUFSZ);
        async_cp16(gB + kt + 8, lB + buf * BUFSZ + 8);
    };
    const int nT = K >> 5;
    issue(0, 0);
    for (int t = 0; t < nT; ++t) {
        const int cur = t & 1;
        if (t + 1 < nT) {
            issue((t + 1) << 5, (t + 1) & 1);
            wait_async4();   // batch t landed (per-wave in-order ASYNCcnt)
        } else {
            wait_async0();
        }
        __syncthreads();
        compute_tile(&sA[cur][0], &sB[cur][0], waveM, waveN, h, ln, acc);
        __syncthreads();
    }
#else
    for (int kt = 0; kt < K; kt += 32) {
        uint4 qa0 = *(const uint4*)(gA + kt);
        uint4 qa1 = *(const uint4*)(gA + kt + 8);
        uint4 qb0 = *(const uint4*)(gB + kt);
        uint4 qb1 = *(const uint4*)(gB + kt + 8);
        __syncthreads();   // previous compute done before overwrite
        ((uint4*)lA)[0] = qa0;
        *(uint4*)(lA + 8) = qa1;
        ((uint4*)lB)[0] = qb0;
        *(uint4*)(lB + 8) = qb1;
        __syncthreads();
        compute_tile(&sA[0][0], &sB[0][0], waveM, waveN, h, ln, acc);
    }
#endif

    // ---- epilogue ----
    for (int i = 0; i < 2; ++i) {
        for (int j = 0; j < 4; ++j) {
            for (int r = 0; r < 8; ++r) {
                const int lm = waveM * 32 + i * 16 + 8 * h + r;  // local row
                const int lc = waveN * 64 + j * 16 + ln;         // local col
                float v = acc[i][j][r];
                if (EPI == 0) {
                    const int gn = tileN * 128 + lc;
                    float o = v + bias[gn];
                    if (o < 0.f) o = 0.f;
                    outB[(size_t)sRows[lm] * (size_t)N + gn] = f2bf(o);
                } else if (EPI == 1) {
                    const int gn = tileN * 128 + lc;             // 0..511
                    const int s  = sRows[lm];
                    const int id = yv[s] * 2 + ev[s];
                    if ((gn >> 7) == id)
                        outF[(size_t)s * D2 + (gn & 127)] = v + bias[gn];
                } else { // EPI == 2
                    const bool valid = (rowTile * 128 + lm) < cnt;
                    const int t = tileN * 128 + lc;
                    if (valid && t < NT) {
                        const int s = sRows[lm];
                        int rr = (int)((sqrtf(8.0f * (float)t + 1.0f) - 1.0f) * 0.5f);
                        while ((rr + 1) * (rr + 2) / 2 <= t) ++rr;
                        while (rr * (rr + 1) / 2 > t) --rr;
                        const int cc = t - rr * (rr + 1) / 2;
                        float val = v + bias[(size_t)g * NT + t];
                        if (rr == cc)
                            val = (val > 20.f) ? val : log1pf(expf(val));
                        outF[(size_t)s * (D2 * D2) + (size_t)rr * D2 + cc] = val;
                    }
                }
            }
        }
    }
}

// ---------------- host side ----------------

static inline size_t alignUp(size_t x, size_t a) { return (x + a - 1) & ~(a - 1); }

extern "C" void kernel_launch(void* const* d_in, const int* in_sizes, int n_in,
                              void* d_out, int out_size, void* d_ws, size_t ws_size,
                              hipStream_t stream) {
    (void)in_sizes; (void)n_in; (void)out_size; (void)ws_size;
    const float* x    = (const float*)d_in[0];
    const int*   y    = (const int*)  d_in[1];
    const int*   e    = (const int*)  d_in[2];
    const float* W1   = (const float*)d_in[3];
    const float* b1   = (const float*)d_in[4];
    const float* W2   = (const float*)d_in[5];
    const float* b2   = (const float*)d_in[6];
    const float* Wmu  = (const float*)d_in[7];
    const float* bmu  = (const float*)d_in[8];
    const float* Wcov = (const float*)d_in[9];
    const float* bcov = (const float*)d_in[10];

    float* outMu = (float*)d_out;                       // [4096,128]
    float* outL  = (float*)d_out + (size_t)B_SAMP * D2; // [4096,128,128]

    char* ws = (char*)d_ws;
    size_t off = 0;
    u16* xb    = (u16*)(ws + off); off = alignUp(off + (size_t)B_SAMP * XDIM * 2, 256);
    u16* w1t   = (u16*)(ws + off); off = alignUp(off + (size_t)XDIM * HDIM * 2, 256);
    u16* w2t   = (u16*)(ws + off); off = alignUp(off + (size_t)HDIM * HDIM * 2, 256);
    u16* wmut  = (u16*)(ws + off); off = alignUp(off + (size_t)HDIM * MU_N * 2, 256);
    u16* wcovt = (u16*)(ws + off); off = alignUp(off + (size_t)HDIM * COV_N * 2, 256);
    u16* h1b   = (u16*)(ws + off); off = alignUp(off + (size_t)B_SAMP * HDIM * 2, 256);
    u16* h2b   = (u16*)(ws + off); off = alignUp(off + (size_t)B_SAMP * HDIM * 2, 256);
    int* gcnt  = (int*)(ws + off); off = alignUp(off + 4 * sizeof(int), 256);
    int* grows = (int*)(ws + off); off = alignUp(off + 4 * (size_t)B_SAMP * sizeof(int), 256);

    const int T = 256;
    auto blocks = [](long n, int t) { return (unsigned)((n + t - 1) / t); };

    // 1) activations: plain f32->bf16 ; weights: fused transpose+convert
    cvt_f32_bf16<<<blocks((long)B_SAMP * XDIM, T), T, 0, stream>>>(x, xb, (long)B_SAMP * XDIM);
    cvt_transpose_bf16<<<dim3(HDIM / 32, XDIM / 32), T, 0, stream>>>(W1, w1t, XDIM, HDIM);
    cvt_transpose_bf16<<<dim3(HDIM / 32, HDIM / 32), T, 0, stream>>>(W2, w2t, HDIM, HDIM);
    cvt_transpose_bf16<<<dim3(MU_N / 32, HDIM / 32), T, 0, stream>>>(Wmu, wmut, HDIM, MU_N);
    cvt_transpose_bf16<<<dim3(COV_N / 32, HDIM / 32), T, 0, stream>>>(Wcov, wcovt, HDIM, COV_N);

    // 2) group samples by idx = y*2+e
    zero_cnt<<<1, 32, 0, stream>>>(gcnt);
    build_groups<<<blocks(B_SAMP, T), T, 0, stream>>>(y, e, gcnt, grows);

    // 3) zero L (upper triangle stays 0)
    long Lfloat4 = (long)B_SAMP * D2 * D2 / 4;
    zero_f32v4<<<blocks(Lfloat4, T), T, 0, stream>>>(outL, Lfloat4);

    // 4) trunk
    gemm_wmma<0><<<dim3(HDIM / 128, B_SAMP / 128), T, 0, stream>>>(
        xb, w1t, b1, h1b, nullptr, B_SAMP, HDIM, XDIM,
        nullptr, nullptr, nullptr, nullptr);
    gemm_wmma<0><<<dim3(HDIM / 128, B_SAMP / 128), T, 0, stream>>>(
        h1b, w2t, b2, h2b, nullptr, B_SAMP, HDIM, HDIM,
        nullptr, nullptr, nullptr, nullptr);

    // 5) mu head
    gemm_wmma<1><<<dim3(MU_N / 128, B_SAMP / 128), T, 0, stream>>>(
        h2b, wmut, bmu, nullptr, outMu, B_SAMP, MU_N, HDIM,
        y, e, nullptr, nullptr);

    // 6) cov head: grouped rows, fused tril scatter + softplus diag
    gemm_wmma<2><<<dim3((NT + 127) / 128, 4 * 32), T, 0, stream>>>(
        h2b, wcovt, bcov, nullptr, outL, B_SAMP, NT, HDIM,
        nullptr, nullptr, gcnt, grows);
}
